// TransformerBlock_80719615361623
// MI455X (gfx1250) — matrix operand order repaired and checked by reading, not compiled, not run on hardware
//
#include <hip/hip_runtime.h>
#include <hip/hip_bf16.h>

// ---------------------------------------------------------------------------
// Types / helpers
// ---------------------------------------------------------------------------
typedef __attribute__((ext_vector_type(16))) __bf16 v16bf;
typedef __attribute__((ext_vector_type(8)))  __bf16 v8bf;
typedef __attribute__((ext_vector_type(8)))  float  v8f;

#define WMMA_BF16(A, B, C)                                                     \
  __builtin_amdgcn_wmma_f32_16x16x32_bf16(false, (A), false, (B), (short)0,    \
                                          (C), false, false)

__device__ __forceinline__ __bf16 f2bf(float f) {
  unsigned u = __builtin_bit_cast(unsigned, f);
  unsigned r = u + 0x7FFFu + ((u >> 16) & 1u); // round-to-nearest-even
  unsigned short h = (unsigned short)(r >> 16);
  return __builtin_bit_cast(__bf16, h);
}
__device__ __forceinline__ float bf2f(__bf16 h) {
  unsigned u = ((unsigned)__builtin_bit_cast(unsigned short, h)) << 16;
  return __builtin_bit_cast(float, u);
}
__device__ __forceinline__ v8f zero8() {
  v8f z;
#pragma unroll
  for (int i = 0; i < 8; ++i) z[i] = 0.0f;
  return z;
}

// DPP16 row_ror cross-lane move (rotation within each 16-lane row).
template <int ROR>
__device__ __forceinline__ float dpp_ror(float x) {
  return __builtin_bit_cast(
      float, __builtin_amdgcn_update_dpp(0, __builtin_bit_cast(int, x),
                                         0x120 | ROR, 0xf, 0xf, true));
}
// Butterfly over a 16-lane row: rotations 1,2,4,8 generate the full group, so
// every lane ends with the row-wide reduction (valid for max and sum).
__device__ __forceinline__ float rowmax16(float t) {
  t = fmaxf(t, dpp_ror<1>(t));
  t = fmaxf(t, dpp_ror<2>(t));
  t = fmaxf(t, dpp_ror<4>(t));
  t = fmaxf(t, dpp_ror<8>(t));
  return t;
}
__device__ __forceinline__ float rowsum16(float t) {
  t += dpp_ror<1>(t);
  t += dpp_ror<2>(t);
  t += dpp_ror<4>(t);
  t += dpp_ror<8>(t);
  return t;
}

// A fragment, 16x32 bf16 (swizzled K). Caller passes p = row_base + kk + ((lane>>4)<<3).
__device__ __forceinline__ v16bf load_a(const __bf16* p) {
  v8bf lo = *(const v8bf*)p;
  v8bf hi = *(const v8bf*)(p + 16);
  return __builtin_shufflevector(lo, hi, 0, 1, 2, 3, 4, 5, 6, 7, 8, 9, 10, 11,
                                 12, 13, 14, 15);
}
// B fragment, 32x16 bf16 (swizzled K, same as A). p = col_base + kk + ((lane>>4)<<3).
__device__ __forceinline__ v16bf load_b(const __bf16* p) {
  v8bf lo = *(const v8bf*)p;
  v8bf hi = *(const v8bf*)(p + 16);
  return __builtin_shufflevector(lo, hi, 0, 1, 2, 3, 4, 5, 6, 7, 8, 9, 10, 11,
                                 12, 13, 14, 15);
}

// ---------------------------------------------------------------------------
// fp32 -> bf16 conversion (weights)
// ---------------------------------------------------------------------------
__global__ __launch_bounds__(256) void f32_to_bf16_kernel(
    const float* __restrict__ src, __bf16* __restrict__ dst, int n) {
  int i = blockIdx.x * 256 + threadIdx.x;
  if (i < n) dst[i] = f2bf(src[i]);
}

// ---------------------------------------------------------------------------
// LayerNorm (D = 1024), fp32 in -> bf16 out. One block (256 threads) per row.
// ---------------------------------------------------------------------------
__global__ __launch_bounds__(256) void layernorm_bf16_kernel(
    const float* __restrict__ x, const float* __restrict__ w,
    const float* __restrict__ b, __bf16* __restrict__ out) {
  const int row = blockIdx.x;
  const int tid = threadIdx.x;
  const float4 f = ((const float4*)(x + (size_t)row * 1024))[tid];
  float s = f.x + f.y + f.z + f.w;
  float s2 = f.x * f.x + f.y * f.y + f.z * f.z + f.w * f.w;
#pragma unroll
  for (int d = 16; d >= 1; d >>= 1) {
    s += __shfl_xor(s, d, 32);
    s2 += __shfl_xor(s2, d, 32);
  }
  __shared__ float sh_s[8], sh_s2[8];
  if ((tid & 31) == 0) {
    sh_s[tid >> 5] = s;
    sh_s2[tid >> 5] = s2;
  }
  __syncthreads();
  s = 0.0f;
  s2 = 0.0f;
#pragma unroll
  for (int i = 0; i < 8; ++i) {
    s += sh_s[i];
    s2 += sh_s2[i];
  }
  const float mean = s * (1.0f / 1024.0f);
  const float var = s2 * (1.0f / 1024.0f) - mean * mean;
  const float inv = rsqrtf(var + 1e-5f);
  const float4 wv = ((const float4*)w)[tid];
  const float4 bv = ((const float4*)b)[tid];
  __bf16* o = out + (size_t)row * 1024 + tid * 4;
  o[0] = f2bf((f.x - mean) * inv * wv.x + bv.x);
  o[1] = f2bf((f.y - mean) * inv * wv.y + bv.y);
  o[2] = f2bf((f.z - mean) * inv * wv.z + bv.z);
  o[3] = f2bf((f.w - mean) * inv * wv.w + bv.w);
}

// ---------------------------------------------------------------------------
// WMMA GEMM: out[M,N] = A[M,K](bf16) * W[N,K]^T(bf16) + bias (+ epilogue)
// Wave tile 64x64 (16 WMMAs / 8 fragment loads per K-step), block = 2x2 waves
// -> 128x128 block tile. Grid: (N/128, M/128).
// EPI: 0 = bias, store bf16;  1 = bias + fp32 residual, store fp32;
//      2 = bias + exact GELU, store bf16
// ---------------------------------------------------------------------------
template <int EPI>
__global__ __launch_bounds__(128) void gemm_bf16_kernel(
    const __bf16* __restrict__ A, const __bf16* __restrict__ W,
    const float* __restrict__ bias, const float* __restrict__ res,
    void* __restrict__ outp, int M, int N, int K) {
  const int lane = threadIdx.x & 31;
  const int warp = threadIdx.x >> 5;
  const int lm = lane & 15;
  const int lka = (lane >> 4) << 3;
  const int lkb = (lane >> 4) << 3;
  const int n0 = blockIdx.x * 128 + (warp & 1) * 64;
  const int m0 = blockIdx.y * 128 + (warp >> 1) * 64;

  const __bf16* ap = A + (size_t)(m0 + lm) * K + lka;
  const __bf16* bp = W + (size_t)(n0 + lm) * K + lkb;
  const size_t step16 = (size_t)16 * K;

  v8f acc[4][4];
#pragma unroll
  for (int i = 0; i < 4; ++i)
#pragma unroll
    for (int j = 0; j < 4; ++j) acc[i][j] = zero8();

  for (int kk = 0; kk < K; kk += 32) {
    v16bf a[4], b[4];
#pragma unroll
    for (int i = 0; i < 4; ++i) a[i] = load_a(ap + (size_t)i * step16 + kk);
#pragma unroll
    for (int j = 0; j < 4; ++j) b[j] = load_b(bp + (size_t)j * step16 + kk);
#pragma unroll
    for (int i = 0; i < 4; ++i)
#pragma unroll
      for (int j = 0; j < 4; ++j) acc[i][j] = WMMA_BF16(a[i], b[j], acc[i][j]);
  }

  const int rbase = m0 + 8 * (lane >> 4);
#pragma unroll
  for (int j = 0; j < 4; ++j) {
    const int col = n0 + j * 16 + lm;
    const float bv = bias[col];
#pragma unroll
    for (int i = 0; i < 4; ++i) {
#pragma unroll
      for (int v = 0; v < 8; ++v) {
        const size_t idx = (size_t)(rbase + i * 16 + v) * (size_t)N + col;
        float val = acc[i][j][v] + bv;
        if constexpr (EPI == 0) {
          ((__bf16*)outp)[idx] = f2bf(val);
        } else if constexpr (EPI == 1) {
          ((float*)outp)[idx] = val + res[idx];
        } else {
          ((__bf16*)outp)[idx] =
              f2bf(0.5f * val * (1.0f + erff(val * 0.70710678118f)));
        }
      }
    }
  }
}

// ---------------------------------------------------------------------------
// Rearrange qkv (B,S,3,H,64) bf16 -> q (B,H,S,64) *0.125, k (B,H,S,64),
// vT (B,H,64,S)
// ---------------------------------------------------------------------------
__global__ __launch_bounds__(256) void qkv_rearrange_kernel(
    const __bf16* __restrict__ qkv, __bf16* __restrict__ q,
    __bf16* __restrict__ k, __bf16* __restrict__ vT) {
  const int idx = blockIdx.x * 256 + threadIdx.x; // < 8*1024*16*64 = 2^23
  const int d = idx & 63;
  const int h = (idx >> 6) & 15;
  const int s = (idx >> 10) & 1023;
  const int b = idx >> 20;
  const size_t base = (size_t)(b * 1024 + s) * 3072;
  const __bf16 qv = qkv[base + h * 64 + d];
  const __bf16 kv = qkv[base + 1024 + h * 64 + d];
  const __bf16 vv = qkv[base + 2048 + h * 64 + d];
  const size_t bh = (size_t)(b * 16 + h);
  q[(bh * 1024 + s) * 64 + d] = f2bf(bf2f(qv) * 0.125f); // 1/sqrt(64)
  k[(bh * 1024 + s) * 64 + d] = kv;
  vT[(bh * 64 + d) * 1024 + s] = vv;
}

// ---------------------------------------------------------------------------
// Flash attention: wave owns 16 q rows of one (b,h); loops 1024 keys by 32.
// scores: 4 WMMAs, online softmax with DPP16 row butterflies (pure VALU),
// P staged via LDS (s_wait_dscnt) into A-fragment layout, then P@V: 4 WMMAs.
// Grid: 2048 blocks x 128 threads (4 waves).
// ---------------------------------------------------------------------------
__global__ __launch_bounds__(128) void attention_kernel(
    const __bf16* __restrict__ q, const __bf16* __restrict__ kbuf,
    const __bf16* __restrict__ vT, __bf16* __restrict__ o) {
  __shared__ __attribute__((aligned(16))) __bf16 pls[4][16 * 32];
  const int lane = threadIdx.x & 31;
  const int warp = threadIdx.x >> 5;
  const int wid = blockIdx.x * 4 + warp; // 0..8191
  const int q0 = (wid & 63) * 16;        // q-row tile within (b,h)
  const int bh = wid >> 6;               // 0..127
  const int b = bh >> 4, h = bh & 15;
  const int lm = lane & 15;
  const int lka = (lane >> 4) << 3;
  const int lkb = (lane >> 4) << 3;

  const __bf16* qb = q + (size_t)bh * 1024 * 64;
  const __bf16* kb = kbuf + (size_t)bh * 1024 * 64;
  const __bf16* vb = vT + (size_t)bh * 64 * 1024;

  // q A-fragments over head-dim (K = 64 -> two 16x32 fragments)
  const __bf16* qp = qb + (size_t)(q0 + lm) * 64 + lka;
  const v16bf qa0 = load_a(qp);
  const v16bf qa1 = load_a(qp + 32);

  float mrow[8], lrow[8];
  v8f oacc[4];
#pragma unroll
  for (int v = 0; v < 8; ++v) {
    mrow[v] = -1e30f;
    lrow[v] = 0.0f;
  }
#pragma unroll
  for (int nt = 0; nt < 4; ++nt) oacc[nt] = zero8();

  __bf16* pw = &pls[warp][0];

  for (int kt = 0; kt < 1024; kt += 32) {
    // K^T B-fragments: col = key index, k = head-dim
    const __bf16* kp = kb + (size_t)(kt + lm) * 64 + lkb;
    v16bf k0a = load_b(kp);
    v16bf k0b = load_b(kp + 32);
    v16bf k1a = load_b(kp + 16 * 64);
    v16bf k1b = load_b(kp + 16 * 64 + 32);

    v8f s0 = zero8();
    s0 = WMMA_BF16(qa0, k0a, s0);
    s0 = WMMA_BF16(qa1, k0b, s0);
    v8f s1 = zero8();
    s1 = WMMA_BF16(qa0, k1a, s1);
    s1 = WMMA_BF16(qa1, k1b, s1);

    // online softmax update (rows v + 8*(lane>>4), replicated over 16 lanes)
    float alpha[8];
    v8f p0, p1;
#pragma unroll
    for (int v = 0; v < 8; ++v) {
      const float t = rowmax16(fmaxf(s0[v], s1[v]));
      const float mnew = fmaxf(mrow[v], t);
      alpha[v] = __expf(mrow[v] - mnew);
      mrow[v] = mnew;
      const float e0 = __expf(s0[v] - mnew);
      const float e1 = __expf(s1[v] - mnew);
      lrow[v] = lrow[v] * alpha[v] + rowsum16(e0 + e1);
      p0[v] = e0;
      p1[v] = e1;
    }
#pragma unroll
    for (int nt = 0; nt < 4; ++nt)
#pragma unroll
      for (int v = 0; v < 8; ++v) oacc[nt][v] *= alpha[v];

    // stage P (16x32) to LDS in row-major, reload as A-fragment
    {
      const int c = lm;
#pragma unroll
      for (int v = 0; v < 8; ++v) {
        const int r = v + 8 * (lane >> 4);
        pw[r * 32 + c] = f2bf(p0[v]);
        pw[r * 32 + 16 + c] = f2bf(p1[v]);
      }
    }
    asm volatile("s_wait_dscnt 0" ::: "memory");
    const v16bf pa = load_a(pw + lm * 32 + lka);

    // V B-fragments: col = head-dim, k = key (contiguous in vT rows)
    const __bf16* vp = vb + (size_t)lm * 1024 + kt + lkb;
#pragma unroll
    for (int nt = 0; nt < 4; ++nt) {
      v16bf vf = load_b(vp + (size_t)nt * 16 * 1024);
      oacc[nt] = WMMA_BF16(pa, vf, oacc[nt]);
    }
  }

  // normalize and store to o (B,S,DIM) bf16
#pragma unroll
  for (int nt = 0; nt < 4; ++nt) {
#pragma unroll
    for (int v = 0; v < 8; ++v) {
      const int row = q0 + v + 8 * (lane >> 4);
      const int col = h * 64 + nt * 16 + lm;
      o[((size_t)(b * 1024 + row)) * 1024 + col] = f2bf(oacc[nt][v] / lrow[v]);
    }
  }
}

// ---------------------------------------------------------------------------
// Host launcher
// ---------------------------------------------------------------------------
extern "C" void kernel_launch(void* const* d_in, const int* in_sizes, int n_in,
                              void* d_out, int out_size, void* d_ws,
                              size_t ws_size, hipStream_t stream) {
  (void)in_sizes; (void)n_in; (void)out_size; (void)ws_size;
  const float* x    = (const float*)d_in[0];
  const float* ln1w = (const float*)d_in[1];
  const float* ln1b = (const float*)d_in[2];
  const float* qkvw = (const float*)d_in[3];
  const float* qkvb = (const float*)d_in[4];
  const float* outw = (const float*)d_in[5];
  const float* outb = (const float*)d_in[6];
  const float* ln2w = (const float*)d_in[7];
  const float* ln2b = (const float*)d_in[8];
  const float* fc1w = (const float*)d_in[9];
  const float* fc1b = (const float*)d_in[10];
  const float* fc2w = (const float*)d_in[11];
  const float* fc2b = (const float*)d_in[12];

  char* ws = (char*)d_ws;
  __bf16* wq  = (__bf16*)(ws + 0);          // 3072*1024 bf16
  __bf16* wo  = (__bf16*)(ws + 6291456);    // 1024*1024
  __bf16* wf1 = (__bf16*)(ws + 8388608);    // 4096*1024
  __bf16* wf2 = (__bf16*)(ws + 16777216);   // 1024*4096
  __bf16* h1  = (__bf16*)(ws + 25165824);   // 8192*1024 (reused as h2)
  float*  x1  = (float*) (ws + 41943040);   // 8192*1024 fp32
  __bf16* qkv = (__bf16*)(ws + 75497472);   // 8192*3072 (reused as hh 8192*4096)
  __bf16* qb  = (__bf16*)(ws + 125829120);  // (B,H,S,64)
  __bf16* kb  = (__bf16*)(ws + 142606336);  // (B,H,S,64)
  __bf16* vT  = (__bf16*)(ws + 159383552);  // (B,H,64,S)
  __bf16* ob  = (__bf16*)(ws + 176160768);  // 8192*1024

  // weight conversion
  f32_to_bf16_kernel<<<3145728 / 256, 256, 0, stream>>>(qkvw, wq, 3145728);
  f32_to_bf16_kernel<<<1048576 / 256, 256, 0, stream>>>(outw, wo, 1048576);
  f32_to_bf16_kernel<<<4194304 / 256, 256, 0, stream>>>(fc1w, wf1, 4194304);
  f32_to_bf16_kernel<<<4194304 / 256, 256, 0, stream>>>(fc2w, wf2, 4194304);

  // LN1 -> qkv GEMM -> rearrange -> attention
  layernorm_bf16_kernel<<<8192, 256, 0, stream>>>(x, ln1w, ln1b, h1);
  gemm_bf16_kernel<0><<<dim3(24, 64), 128, 0, stream>>>(h1, wq, qkvb, nullptr,
                                                        qkv, 8192, 3072, 1024);
  qkv_rearrange_kernel<<<8388608 / 256, 256, 0, stream>>>(qkv, qb, kb, vT);
  attention_kernel<<<2048, 128, 0, stream>>>(qb, kb, vT, ob);

  // out-proj + residual -> x1
  gemm_bf16_kernel<1><<<dim3(8, 64), 128, 0, stream>>>(ob, wo, outb, x, x1,
                                                       8192, 1024, 1024);
  // LN2 -> FC1(+GELU) -> FC2 + residual -> d_out
  layernorm_bf16_kernel<<<8192, 256, 0, stream>>>(x1, ln2w, ln2b, h1);
  __bf16* hh = qkv;
  gemm_bf16_kernel<2><<<dim3(32, 64), 128, 0, stream>>>(h1, wf1, fc1b, nullptr,
                                                        hh, 8192, 4096, 1024);
  gemm_bf16_kernel<1><<<dim3(8, 64), 128, 0, stream>>>(
      hh, wf2, fc2b, x1, (float*)d_out, 8192, 1024, 4096);
}
